// HybridChatbotModel_25280177504838
// MI455X (gfx1250) — matrix operand order repaired and checked by reading, not compiled, not run on hardware
//
#include <hip/hip_runtime.h>
#include <math.h>

// ---------------------------------------------------------------------------
// HybridChatbotModel forward for MI455X (gfx1250, wave32, WMMA).
// All large matmuls use v_wmma_f32_16x16x32_bf16 (f32 accumulate), operands
// converted f32->bf16 with the hardware cvt while staging to LDS (double-
// buffered). LSTM recurrence runs as one persistent 16-wave workgroup per
// direction with LDS-resident h/c state; its Whh weights are pre-converted to
// bf16 once per launch and re-streamed from L2 every timestep (a compiler
// memory barrier stops LICM from hoisting + spilling the fragments).
// ---------------------------------------------------------------------------

#define B_  4
#define S_  512
#define V_  32000
#define E_  256
#define H_  512
#define D_  1024
#define NH_ 8
#define NL_ 6
#define BS_ (B_ * S_)

typedef __attribute__((ext_vector_type(16))) __bf16 v16bf;
typedef __attribute__((ext_vector_type(8)))  __bf16 v8bf;
typedef __attribute__((ext_vector_type(4)))  __bf16 v4bf;
typedef __attribute__((ext_vector_type(8)))  float  v8f;

// float -> bf16 via hardware convert (v_cvt_pk_bf16_f32 on gfx1250)
__device__ __forceinline__ __bf16 f2bf(float f) { return (__bf16)f; }

__device__ __forceinline__ v8f wmma_bf16(v16bf a, v16bf b, v8f c) {
  // 8 args: (neg_a, A, neg_b, B, c_mod, C, reuse_a, reuse_b)
  return __builtin_amdgcn_wmma_f32_16x16x32_bf16(false, a, false, b, (short)0, c,
                                                 false, false);
}

// A-fragment (16x32 bf16, M x K) from LDS laid out [row][k], row stride `stride`.
// Contiguous k order (same as frag_b_lds): lanes 0-15 hold row M=lane with K k0..k0+15,
// lanes 16-31 hold row M=lane-16 with K k0+16..k0+31.
__device__ __forceinline__ v16bf frag_a_lds(const __bf16* s, int stride,
                                            int rowBase, int k0, int lane) {
  int row = rowBase + (lane & 15);
  int c   = k0 + ((lane & 16) ? 16 : 0);
  const __bf16* p = s + row * stride + c;
  v8bf lo = *(const v8bf*)p;
  v8bf hi = *(const v8bf*)(p + 8);
  return __builtin_shufflevector(lo, hi, 0,1,2,3,4,5,6,7,8,9,10,11,12,13,14,15);
}

// B-fragment (32x16 bf16, K x N) from LDS laid out [n][k]:
// lanes 0-15: col N=lane, K k0..k0+15; lanes 16-31: col N=lane-16, K k0+16..k0+31.
__device__ __forceinline__ v16bf frag_b_lds(const __bf16* s, int stride,
                                            int rowBase, int k0, int lane) {
  int row = rowBase + (lane & 15);
  int c   = k0 + ((lane & 16) ? 16 : 0);
  return *(const v16bf*)(s + row * stride + c);
}

// B-fragment directly from a pre-converted global bf16 [N][ld] matrix
// (LSTM recurrence; Whh_bf16 is L2-resident). One 32B load per lane.
__device__ __forceinline__ v16bf frag_b_gbf16(const __bf16* W, int ld,
                                              int nBase, int k0, int lane) {
  const __bf16* p = W + (long)(nBase + (lane & 15)) * ld + k0 + ((lane & 16) ? 16 : 0);
  return *(const v16bf*)p;
}

// ---------------------------------------------------------------------------
// Batched GEMM:  C = act( scale * A @ op(B) + bias )
//   TRANSB=1: B is [N][ldb] row-major  (C = A @ B^T, the x@W.T pattern, Q@K^T)
//   TRANSB=0: B is [K][ldb] row-major  (C = A @ B, the softmax@V pattern)
// Block tile 128x128, K-step 32, double-buffered LDS staging. 8 waves
// (4 along M x 2 along N), each wave owns 32x64 = 2x4 wmma accumulators.
// All dims are multiples of tile sizes (M%128, N%128, K%32 == 0): no bounds.
// z batching: operand += (z/zInner)*zOuterStride + (z%zInner)*zInnerStride.
// ---------------------------------------------------------------------------
#define LDSB 40   // padded K-stride (elements) -> 80B rows, 16B aligned frags

template <int TRANSB>
__global__ __launch_bounds__(256) void gemm_wmma(
    const float* __restrict__ A, const float* __restrict__ Bm,
    const float* __restrict__ bias, float* __restrict__ C,
    int M, int N, int K, int lda, int ldb, int ldc,
    float scale, int relu, int zInner,
    long zAo, long zAi, long zBo, long zBi, long zCo, long zCi) {
  __shared__ __bf16 sA[2][128 * LDSB];
  __shared__ __bf16 sB[2][128 * LDSB];

  int z = blockIdx.z;
  int zo = z / zInner, zi = z - zo * zInner;
  A  += zo * zAo + zi * zAi;
  Bm += zo * zBo + zi * zBi;
  C  += zo * zCo + zi * zCi;

  const int bm = blockIdx.y * 128;
  const int bn = blockIdx.x * 128;
  const int tid = threadIdx.x;
  const int lane = tid & 31;
  const int wid = tid >> 5;
  const int wm = (wid & 3) * 32;   // wave row base within block
  const int wn = (wid >> 2) * 64;  // wave col base within block

  // Stage one 32-deep K tile of A and B into LDS buffer `buf` (f32 -> bf16).
  auto stage = [&](int buf, int k0) {
    float4 fa[4], fb[4];
#pragma unroll
    for (int it = 0; it < 4; ++it) {
      int idx = tid + it * 256;          // 0..1023
      int r = idx >> 3;                  // 0..127
      int c4 = (idx & 7) * 4;            // 0..28
      fa[it] = *(const float4*)(A + (long)(bm + r) * lda + k0 + c4);
    }
    if (TRANSB) {
#pragma unroll
      for (int it = 0; it < 4; ++it) {
        int idx = tid + it * 256;
        int r = idx >> 3, c4 = (idx & 7) * 4;
        fb[it] = *(const float4*)(Bm + (long)(bn + r) * ldb + k0 + c4);
      }
    } else {
#pragma unroll
      for (int it = 0; it < 4; ++it) {
        int idx = tid + it * 256;
        int kk = idx >> 5;               // 0..31
        int n4 = (idx & 31) * 4;         // 0..124
        fb[it] = *(const float4*)(Bm + (long)(k0 + kk) * ldb + bn + n4);
      }
    }
#pragma unroll
    for (int it = 0; it < 4; ++it) {
      int idx = tid + it * 256;
      int r = idx >> 3, c4 = (idx & 7) * 4;
      __bf16* sp = sA[buf] + r * LDSB + c4;
      sp[0] = f2bf(fa[it].x); sp[1] = f2bf(fa[it].y);
      sp[2] = f2bf(fa[it].z); sp[3] = f2bf(fa[it].w);
    }
    if (TRANSB) {
#pragma unroll
      for (int it = 0; it < 4; ++it) {
        int idx = tid + it * 256;
        int r = idx >> 3, c4 = (idx & 7) * 4;
        __bf16* sp = sB[buf] + r * LDSB + c4;
        sp[0] = f2bf(fb[it].x); sp[1] = f2bf(fb[it].y);
        sp[2] = f2bf(fb[it].z); sp[3] = f2bf(fb[it].w);
      }
    } else {
#pragma unroll
      for (int it = 0; it < 4; ++it) {
        int idx = tid + it * 256;
        int kk = idx >> 5, n4 = (idx & 31) * 4;
        sB[buf][(n4 + 0) * LDSB + kk] = f2bf(fb[it].x);
        sB[buf][(n4 + 1) * LDSB + kk] = f2bf(fb[it].y);
        sB[buf][(n4 + 2) * LDSB + kk] = f2bf(fb[it].z);
        sB[buf][(n4 + 3) * LDSB + kk] = f2bf(fb[it].w);
      }
    }
  };

  v8f zero = {0.f, 0.f, 0.f, 0.f, 0.f, 0.f, 0.f, 0.f};
  v8f acc[2][4];
#pragma unroll
  for (int i = 0; i < 2; ++i)
#pragma unroll
    for (int j = 0; j < 4; ++j) acc[i][j] = zero;

  stage(0, 0);
  int cur = 0;
  for (int k0 = 0; k0 < K; k0 += 32) {
    __syncthreads();
    if (k0 + 32 < K) stage(cur ^ 1, k0 + 32);   // prefetch next tile

    v16bf a0 = frag_a_lds(sA[cur], LDSB, wm, 0, lane);
    v16bf a1 = frag_a_lds(sA[cur], LDSB, wm + 16, 0, lane);
#pragma unroll
    for (int nt = 0; nt < 4; ++nt) {
      v16bf bf = frag_b_lds(sB[cur], LDSB, wn + nt * 16, 0, lane);
      acc[0][nt] = wmma_bf16(a0, bf, acc[0][nt]);
      acc[1][nt] = wmma_bf16(a1, bf, acc[1][nt]);
    }
    cur ^= 1;
  }

  // ---- epilogue: C layout (lane 0-15: N=lane, M=r; lane 16-31: N=lane-16, M=r+8)
  const int halo = (lane & 16) ? 8 : 0;
  const int cn = lane & 15;
#pragma unroll
  for (int mt = 0; mt < 2; ++mt) {
#pragma unroll
    for (int nt = 0; nt < 4; ++nt) {
      int col = bn + wn + nt * 16 + cn;
      float bv = bias ? bias[col] : 0.f;
#pragma unroll
      for (int r = 0; r < 8; ++r) {
        int row = bm + wm + mt * 16 + r + halo;
        float v = acc[mt][nt][r] * scale + bv;
        if (relu) v = fmaxf(v, 0.f);
        C[(long)row * ldc + col] = v;
      }
    }
  }
}

// ---------------------------------------------------------------------------
// f32 -> bf16 bulk conversion (for LSTM Whh weights, once per launch).
// n must be a multiple of 1024; each thread converts 4 elements (b64 store).
// ---------------------------------------------------------------------------
__global__ __launch_bounds__(256) void cvt_bf16_kernel(
    const float* __restrict__ in, __bf16* __restrict__ out, long n) {
  long i = (long)blockIdx.x * 1024 + (long)threadIdx.x * 4;
  if (i + 3 < n) {
    float4 f = *(const float4*)(in + i);
    v4bf o;
    o[0] = f2bf(f.x); o[1] = f2bf(f.y); o[2] = f2bf(f.z); o[3] = f2bf(f.w);
    *(v4bf*)(out + i) = o;
  }
}

// ---------------------------------------------------------------------------
// Persistent bi-LSTM layer kernel. blockIdx.x = direction (1 = reversed).
// pre : [2][BS][4H] precomputed x@Wih^T + b.  Whh : [2][4H][H] bf16 (L2 resident).
// hOut: [B][S][2H] (dir d writes columns d*H .. d*H+H-1).
// 512 threads = 16 waves; each wave owns 8 gate-column tiles (128 cols).
// h (bf16) and c (f32) live in LDS across all 512 timesteps; gate GEMM
// (M=4 padded to 16, N=2048, K=512). Whh is re-streamed from L2 each step:
// the asm memory barrier stops LICM from hoisting (and spilling) the
// fragment loads; '#pragma unroll 1' on the K loop bounds live fragments.
// ---------------------------------------------------------------------------
__global__ __launch_bounds__(512) void lstm_layer_kernel(
    const float* __restrict__ pre, const __bf16* __restrict__ Whh,
    float* __restrict__ hOut) {
  const int dir = blockIdx.x;
  pre += (long)dir * BS_ * (4 * H_);
  Whh += (long)dir * (4 * H_) * H_;

  __shared__ __bf16 hS[16 * 520];      // padded: rows 4..15 stay zero
  __shared__ float  cS[4 * H_];
  __shared__ float  gS[4 * (4 * H_)];

  const int tid = threadIdx.x;
  const int lane = tid & 31;
  const int wid = tid >> 5;            // 0..15
  const int n0 = wid * 128;            // 8 gate-column tiles per wave

  for (int i = tid; i < 16 * 520; i += 512) hS[i] = f2bf(0.f);
  for (int i = tid; i < 4 * H_; i += 512) cS[i] = 0.f;
  __syncthreads();

  v8f zero = {0.f, 0.f, 0.f, 0.f, 0.f, 0.f, 0.f, 0.f};

#pragma unroll 1
  for (int t = 0; t < S_; ++t) {
    // Compiler memory barrier: Whh fragment loads must stay inside this loop
    // (they are loop-invariant; without this LLVM hoists ~4KB/lane of
    // fragments out of the loop and spills them all to scratch).
    asm volatile("" ::: "memory");

    const int te = dir ? (S_ - 1 - t) : t;

    // gates = h_prev @ Whh^T
    v8f acc[8];
#pragma unroll
    for (int j = 0; j < 8; ++j) acc[j] = zero;
#pragma unroll 1
    for (int k0 = 0; k0 < H_; k0 += 32) {
      v16bf a = frag_a_lds(hS, 520, 0, k0, lane);
#pragma unroll
      for (int j = 0; j < 8; ++j) {
        v16bf bf = frag_b_gbf16(Whh, H_, n0 + j * 16, k0, lane);
        acc[j] = wmma_bf16(a, bf, acc[j]);
      }
    }
    // valid rows are M=0..3 -> lanes 0-15, accumulator elements 0..3
#pragma unroll
    for (int j = 0; j < 8; ++j) {
      if (lane < 16) {
#pragma unroll
        for (int r = 0; r < 4; ++r)
          gS[r * (4 * H_) + n0 + j * 16 + lane] = acc[j][r];
      }
    }
    __syncthreads();

    // pointwise cell update (4 batches x 512 hidden = 2048 elems, 4 per thread)
#pragma unroll
    for (int it = 0; it < 4; ++it) {
      int idx = tid + it * 512;
      int bm = idx >> 9;
      int hd = idx & (H_ - 1);
      long pb = ((long)bm * S_ + te) * (4 * H_) + hd;
      float gi = gS[bm * (4 * H_) + hd]            + pre[pb];
      float gf = gS[bm * (4 * H_) + H_ + hd]       + pre[pb + H_];
      float gg = gS[bm * (4 * H_) + 2 * H_ + hd]   + pre[pb + 2 * H_];
      float go = gS[bm * (4 * H_) + 3 * H_ + hd]   + pre[pb + 3 * H_];
      gi = 1.f / (1.f + __expf(-gi));
      gf = 1.f / (1.f + __expf(-gf));
      gg = tanhf(gg);
      go = 1.f / (1.f + __expf(-go));
      float c = gf * cS[bm * H_ + hd] + gi * gg;
      cS[bm * H_ + hd] = c;
      float h = go * tanhf(c);
      hS[bm * 520 + hd] = f2bf(h);
      hOut[((long)bm * S_ + te) * (2 * H_) + dir * H_ + hd] = h;
    }
    __syncthreads();
  }
}

// ---------------------------------------------------------------------------
// Pointwise kernels
// ---------------------------------------------------------------------------
__global__ __launch_bounds__(256) void embed_kernel(const int* __restrict__ x,
                                                    const float* __restrict__ emb,
                                                    float* __restrict__ out) {
  int tok = blockIdx.x;
  int e = threadIdx.x;  // E_ == 256
  out[(long)tok * E_ + e] = emb[(long)x[tok] * E_ + e];
}

__device__ __forceinline__ float blk_reduce(float v, float* sbuf, int isMax) {
  int lane = threadIdx.x & 31, wid = threadIdx.x >> 5;
#pragma unroll
  for (int o = 16; o > 0; o >>= 1) {
    float other = __shfl_down(v, o, 32);
    v = isMax ? fmaxf(v, other) : (v + other);
  }
  if (lane == 0) sbuf[wid] = v;
  __syncthreads();
  float r = sbuf[0];
#pragma unroll
  for (int i = 1; i < 8; ++i) r = isMax ? fmaxf(r, sbuf[i]) : (r + sbuf[i]);
  __syncthreads();
  return r;
}

// LayerNorm over D=1024, optional residual input. out may alias x.
__global__ __launch_bounds__(256) void layernorm1024(
    const float* __restrict__ x, const float* __restrict__ res,
    const float* __restrict__ g, const float* __restrict__ b,
    float* __restrict__ out) {
  __shared__ float sbuf[8];
  long row = blockIdx.x;
  const float* xr = x + row * D_;
  const float* rr = res ? res + row * D_ : nullptr;
  int tid = threadIdx.x;
  float v[4], s = 0.f, s2 = 0.f;
#pragma unroll
  for (int i = 0; i < 4; ++i) {
    int idx = tid + i * 256;
    float f = xr[idx] + (rr ? rr[idx] : 0.f);
    v[i] = f; s += f; s2 += f * f;
  }
  s = blk_reduce(s, sbuf, 0);
  s2 = blk_reduce(s2, sbuf, 0);
  float mean = s * (1.f / D_);
  float var = s2 * (1.f / D_) - mean * mean;
  float rstd = rsqrtf(var + 1e-5f);
#pragma unroll
  for (int i = 0; i < 4; ++i) {
    int idx = tid + i * 256;
    out[row * D_ + idx] = (v[i] - mean) * rstd * g[idx] + b[idx];
  }
}

// In-place softmax over rows of width 512.
__global__ __launch_bounds__(256) void softmax512(float* __restrict__ x) {
  __shared__ float sbuf[8];
  long row = blockIdx.x;
  float* p = x + row * 512;
  int tid = threadIdx.x;
  float a = p[tid], b = p[tid + 256];
  float m = blk_reduce(fmaxf(a, b), sbuf, 1);
  float e0 = __expf(a - m), e1 = __expf(b - m);
  float s = blk_reduce(e0 + e1, sbuf, 0);
  float inv = 1.f / s;
  p[tid] = e0 * inv;
  p[tid + 256] = e1 * inv;
}

// ---------------------------------------------------------------------------
// Host-side orchestration
// ---------------------------------------------------------------------------
static void launch_gemm(int transb, const float* A, const float* Bm,
                        const float* bias, float* C, int M, int N, int K,
                        int lda, int ldb, int ldc, float scale, int relu,
                        hipStream_t s, int Z = 1, int zInner = 1,
                        long zAo = 0, long zAi = 0, long zBo = 0, long zBi = 0,
                        long zCo = 0, long zCi = 0) {
  dim3 grid(N / 128, M / 128, Z), blk(256);
  if (transb)
    gemm_wmma<1><<<grid, blk, 0, s>>>(A, Bm, bias, C, M, N, K, lda, ldb, ldc,
                                      scale, relu, zInner, zAo, zAi, zBo, zBi,
                                      zCo, zCi);
  else
    gemm_wmma<0><<<grid, blk, 0, s>>>(A, Bm, bias, C, M, N, K, lda, ldb, ldc,
                                      scale, relu, zInner, zAo, zAi, zBo, zBi,
                                      zCo, zCi);
}

extern "C" void kernel_launch(void* const* d_in, const int* in_sizes, int n_in,
                              void* d_out, int out_size, void* d_ws, size_t ws_size,
                              hipStream_t stream) {
  (void)in_sizes; (void)n_in; (void)out_size; (void)ws_size;
  const int*   x       = (const int*)d_in[0];
  const float* emb     = (const float*)d_in[1];
  const float* Wih0    = (const float*)d_in[2];
  const float* Whh0    = (const float*)d_in[3];
  const float* b0      = (const float*)d_in[4];
  const float* Wih12   = (const float*)d_in[5];
  const float* Whh12   = (const float*)d_in[6];
  const float* b12     = (const float*)d_in[7];
  const float* ln_g    = (const float*)d_in[8];
  const float* ln_b    = (const float*)d_in[9];
  const float* Wqkv    = (const float*)d_in[10];
  const float* bqkv    = (const float*)d_in[11];
  const float* Wo      = (const float*)d_in[12];
  const float* bo      = (const float*)d_in[13];
  const float* ln1g    = (const float*)d_in[14];
  const float* ln1b    = (const float*)d_in[15];
  const float* W1      = (const float*)d_in[16];
  const float* b1      = (const float*)d_in[17];
  const float* W2      = (const float*)d_in[18];
  const float* b2      = (const float*)d_in[19];
  const float* ln2g    = (const float*)d_in[20];
  const float* ln2b    = (const float*)d_in[21];
  const float* attnW   = (const float*)d_in[22];
  const float* attnB   = (const float*)d_in[23];
  const float* outW    = (const float*)d_in[24];
  const float* outBias = (const float*)d_in[25];
  float* logits = (float*)d_out;

  // Workspace layout (floats). FH = one [BS, D] activation = 8MB.
  const long FH = (long)BS_ * D_;
  float* W      = (float*)d_ws;
  float* hA     = W;                              // [BS, D]
  float* hB     = hA + FH;                        // [BS, D]
  float* hT     = hB + FH;                        // [BS, D]
  float* qkv    = hT + FH;                        // [BS, 3D]   (3 FH)
  float* scores = qkv + (long)BS_ * 3 * D_;       // [B*NH,S,S] (4 FH)
  float* ff1    = scores + (long)B_ * NH_ * S_ * S_;  // [BS, 4D] (4 FH)
  float* pre    = scores;                         // reuse: [2][BS][4H] (4 FH)
  float* embB   = ff1;                            // reuse: [BS, E]
  __bf16* whhBf = (__bf16*)(W + 14 * FH);         // [6][4H][H] bf16 (12.6 MB)

  const long WHH1 = (long)2 * 4 * H_ * H_;        // one bi-layer of Whh (2,097,152)
  const float attScale = 0.08838834764831845f;    // 1/sqrt(128)

  // 0) pre-convert all LSTM recurrent weights to bf16 (once per launch)
  cvt_bf16_kernel<<<dim3((unsigned)(WHH1 / 1024)), dim3(256), 0, stream>>>(
      Whh0, whhBf, WHH1);
  cvt_bf16_kernel<<<dim3((unsigned)(2 * WHH1 / 1024)), dim3(256), 0, stream>>>(
      Whh12, whhBf + WHH1, 2 * WHH1);

  // 1) embedding gather
  embed_kernel<<<dim3(BS_), dim3(256), 0, stream>>>(x, emb, embB);

  // 2) LSTM stack (layer 0: E->H, layers 1,2: 2H->H), bidirectional
  for (int d = 0; d < 2; ++d)
    launch_gemm(1, embB, Wih0 + (long)d * 4 * H_ * E_, b0 + (long)d * 4 * H_,
                pre + (long)d * BS_ * 4 * H_, BS_, 4 * H_, E_, E_, E_, 4 * H_,
                1.f, 0, stream);
  lstm_layer_kernel<<<dim3(2), dim3(512), 0, stream>>>(pre, whhBf, hA);

  for (int l = 0; l < 2; ++l) {
    const float* hin = (l == 0) ? hA : hB;
    float* hout = (l == 0) ? hB : hA;
    for (int d = 0; d < 2; ++d) {
      long wi = ((long)l * 2 + d);
      launch_gemm(1, hin, Wih12 + wi * 4 * H_ * D_, b12 + wi * 4 * H_,
                  pre + (long)d * BS_ * 4 * H_, BS_, 4 * H_, D_, D_, D_, 4 * H_,
                  1.f, 0, stream);
    }
    lstm_layer_kernel<<<dim3(2), dim3(512), 0, stream>>>(
        pre, whhBf + (1 + l) * WHH1, hout);
  }

  // 3) LayerNorm (in place on hA)
  layernorm1024<<<dim3(BS_), dim3(256), 0, stream>>>(hA, nullptr, ln_g, ln_b, hA);

  // 4) transformer encoder layers (h lives in hA at loop top/bottom)
  for (int l = 0; l < NL_; ++l) {
    launch_gemm(1, hA, Wqkv + (long)l * 3 * D_ * D_, bqkv + (long)l * 3 * D_,
                qkv, BS_, 3 * D_, D_, D_, D_, 3 * D_, 1.f, 0, stream);
    // scores[b,h] = (q @ k^T) / sqrt(dk)
    launch_gemm(1, qkv, qkv + D_, nullptr, scores, S_, S_, D_ / NH_,
                3 * D_, 3 * D_, S_, attScale, 0, stream,
                B_ * NH_, NH_,
                (long)S_ * 3 * D_, 128, (long)S_ * 3 * D_, 128,
                (long)NH_ * S_ * S_, (long)S_ * S_);
    softmax512<<<dim3(B_ * NH_ * S_), dim3(256), 0, stream>>>(scores);
    // attnO[b,:,h*dk:...] = softmax(scores) @ v
    launch_gemm(0, scores, qkv + 2 * D_, nullptr, hB, S_, D_ / NH_, S_,
                S_, 3 * D_, D_, 1.f, 0, stream,
                B_ * NH_, NH_,
                (long)NH_ * S_ * S_, (long)S_ * S_, (long)S_ * 3 * D_, 128,
                (long)S_ * D_, 128);
    launch_gemm(1, hB, Wo + (long)l * D_ * D_, bo + (long)l * D_, hT,
                BS_, D_, D_, D_, D_, D_, 1.f, 0, stream);
    layernorm1024<<<dim3(BS_), dim3(256), 0, stream>>>(
        hA, hT, ln1g + (long)l * D_, ln1b + (long)l * D_, hB);
    launch_gemm(1, hB, W1 + (long)l * 4 * D_ * D_, b1 + (long)l * 4 * D_, ff1,
                BS_, 4 * D_, D_, D_, D_, 4 * D_, 1.f, 1 /*relu*/, stream);
    launch_gemm(1, ff1, W2 + (long)l * D_ * 4 * D_, b2 + (long)l * D_, hT,
                BS_, D_, 4 * D_, 4 * D_, 4 * D_, D_, 1.f, 0, stream);
    layernorm1024<<<dim3(BS_), dim3(256), 0, stream>>>(
        hB, hT, ln2g + (long)l * D_, ln2b + (long)l * D_, hA);
  }

  // 5) final attention block (q,k,v packed into qkv buffer, ldc = 3D)
  for (int p = 0; p < 3; ++p)
    launch_gemm(1, hA, attnW + (long)p * D_ * D_, attnB + (long)p * D_,
                qkv + (long)p * D_, BS_, D_, D_, D_, D_, 3 * D_, 1.f, 0, stream);
  launch_gemm(1, qkv, qkv + D_, nullptr, scores, S_, S_, D_ / NH_,
              3 * D_, 3 * D_, S_, attScale, 0, stream,
              B_ * NH_, NH_,
              (long)S_ * 3 * D_, 128, (long)S_ * 3 * D_, 128,
              (long)NH_ * S_ * S_, (long)S_ * S_);
  softmax512<<<dim3(B_ * NH_ * S_), dim3(256), 0, stream>>>(scores);
  launch_gemm(0, scores, qkv + 2 * D_, nullptr, hB, S_, D_ / NH_, S_,
              S_, 3 * D_, D_, 1.f, 0, stream,
              B_ * NH_, NH_,
              (long)NH_ * S_ * S_, (long)S_ * S_, (long)S_ * 3 * D_, 128,
              (long)S_ * D_, 128);
  launch_gemm(1, hB, attnW + 3L * D_ * D_, attnB + 3L * D_, hT,
              BS_, D_, D_, D_, D_, D_, 1.f, 0, stream);

  // 6) vocab projection -> d_out  [BS, 32000]
  launch_gemm(1, hT, outW, outBias, logits, BS_, V_, D_, D_, D_, V_, 1.f, 0,
              stream);
}
